// GCN_8229157339587
// MI455X (gfx1250) — matrix operand order repaired and checked by reading, not compiled, not run on hardware
//
#include <hip/hip_runtime.h>
#include <math.h>

typedef __attribute__((ext_vector_type(16))) _Float16 v16h;
typedef __attribute__((ext_vector_type(8)))  float    v8f;

#define EPS 1e-5f

__device__ __forceinline__ float relu1(float x) {
  // single v_med3_num_f32: clamp(x, 0, +inf)
  return __builtin_amdgcn_fmed3f(x, 0.0f, __builtin_inff());
}

__device__ __forceinline__ v8f wmma_f16(v16h a, v16h b, v8f c) {
  // (neg_a, A, neg_b, B, c_mod, C, reuse_a, reuse_b)
  return __builtin_amdgcn_wmma_f32_16x16x32_f16(false, a, false, b, (short)0, c,
                                                false, false);
}

// ---------------------------------------------------------------------------
// Stage 1: per-element Linear(2,32)+ReLU, mean over K=8 -> hid0 [N,32]
// ---------------------------------------------------------------------------
__global__ void embed_kernel(const float* __restrict__ feat,
                             const float* __restrict__ W0,  // [2][32]
                             const float* __restrict__ b0,  // [32]
                             float* __restrict__ hid0, int Nn) {
  int i = blockIdx.x * blockDim.x + threadIdx.x;
  if (i >= Nn * 32) return;
  int ch = i & 31;
  int node = i >> 5;
  float w0 = W0[ch], w1 = W0[32 + ch], bb = b0[ch];
  const float* f = feat + node * 16;
  float s = 0.f;
#pragma unroll
  for (int k = 0; k < 8; ++k) {
    s += relu1(f[2 * k] * w0 + f[2 * k + 1] * w1 + bb);
  }
  hid0[i] = s * 0.125f;
}

// ---------------------------------------------------------------------------
// In-degree counts (same for both convs)
// ---------------------------------------------------------------------------
__global__ void count_kernel(const int* __restrict__ dst, float* __restrict__ cnt,
                             int E) {
  int e = blockIdx.x * blockDim.x + threadIdx.x;
  if (e < E) atomicAdd(&cnt[dst[e]], 1.0f);
}

// ---------------------------------------------------------------------------
// EdgeConv main pass: per 16-edge tile per wave:
//   x = [hid[src], hid[dst]] (64), z = relu(x @ W + b) via hi/lo-split f16 WMMA
//   atomically scatter z onto acc[dst], accumulate per-channel sum(z), sum(z^2)
// ---------------------------------------------------------------------------
__global__ void __launch_bounds__(256)
edgeconv_kernel(const float* __restrict__ hid,   // [N,32]
                const int* __restrict__ src, const int* __restrict__ dst,
                const float* __restrict__ W,     // [64][32] row-major
                const float* __restrict__ b,     // [32]
                float* __restrict__ acc,         // [N,32] (pre-zeroed)
                float* __restrict__ gsum,        // [32]   (pre-zeroed)
                float* __restrict__ gssq,        // [32]   (pre-zeroed)
                int E) {
  __shared__ _Float16 Whi[64 * 32];
  __shared__ _Float16 Wlo[64 * 32];
  const int tid = threadIdx.x;
  for (int i = tid; i < 64 * 32; i += blockDim.x) {
    float w = W[i];
    _Float16 h = (_Float16)w;
    Whi[i] = h;
    Wlo[i] = (_Float16)(w - (float)h);
  }
  __syncthreads();

  const int lane = tid & 31;
  const int wave = tid >> 5;
  const int half = lane >> 4;   // which 16-lane half of the wave
  const int n    = lane & 15;   // column / edge-in-tile index

  // --- Build B fragments (32x16 f16, K x N) once; resident all kernel ---
  // Lane layout: lanes 0-15 hold K {0..7,16..23} (col = lane), lanes 16-31 K {8..15,24..31}.
  v16h Bhi[2][2], Blo[2][2];    // [k-chunk][n-tile]
#pragma unroll
  for (int c = 0; c < 2; ++c) {
#pragma unroll
    for (int t = 0; t < 2; ++t) {
      v16h bh, bl;
#pragma unroll
      for (int k = 0; k < 16; ++k) {
        int row = c * 32 + (k & 8) * 2 + half * 8 + (k & 7);
        int col = t * 16 + n;
        bh[k] = Whi[row * 32 + col];
        bl[k] = Wlo[row * 32 + col];
      }
      Bhi[c][t] = bh;
      Blo[c][t] = bl;
    }
  }
  const float bias0 = b[n];
  const float bias1 = b[16 + n];

  float sumz0 = 0.f, ssq0 = 0.f, sumz1 = 0.f, ssq1 = 0.f;

  const int numTiles = (E + 15) >> 4;
  const int gwave  = blockIdx.x * (blockDim.x >> 5) + wave;
  const int nwaves = gridDim.x * (blockDim.x >> 5);

  for (int tile = gwave; tile < numTiles; tile += nwaves) {
    const int base = tile << 4;
    const bool full = (base + 16 <= E);  // wave-uniform
    int e = base + n;
    if (e >= E) e = E - 1;               // clamp; invalid rows masked in tail
    const int s_node = src[e];
    const int d_node = dst[e];

    // --- Gather + hi/lo split into A fragments (16x32 f16, M x K) ---
    // lane(half,n): row M=n; half=0 -> K {0..7,16..23}, half=1 -> {8..15,24..31}
    const float4* ps = (const float4*)hid + s_node * 8;  // 32-bit offset math
    const float4* pd = (const float4*)hid + d_node * 8;
    const int o = half * 2;
    float4 s0 = ps[o], s1 = ps[o + 1], s2 = ps[o + 4], s3 = ps[o + 5];
    float4 t0 = pd[o], t1 = pd[o + 1], t2 = pd[o + 4], t3 = pd[o + 5];

    v16h a0h, a0l, a1h, a1l;
    {
      float xv[16] = {s0.x, s0.y, s0.z, s0.w, s1.x, s1.y, s1.z, s1.w,
                      s2.x, s2.y, s2.z, s2.w, s3.x, s3.y, s3.z, s3.w};
#pragma unroll
      for (int k = 0; k < 16; ++k) {
        _Float16 h = (_Float16)xv[k];
        a0h[k] = h;
        a0l[k] = (_Float16)(xv[k] - (float)h);
      }
    }
    {
      float xv[16] = {t0.x, t0.y, t0.z, t0.w, t1.x, t1.y, t1.z, t1.w,
                      t2.x, t2.y, t2.z, t2.w, t3.x, t3.y, t3.z, t3.w};
#pragma unroll
      for (int k = 0; k < 16; ++k) {
        _Float16 h = (_Float16)xv[k];
        a1h[k] = h;
        a1l[k] = (_Float16)(xv[k] - (float)h);
      }
    }

    // destination row offsets for this lane's 8 C rows (M = half*8 + v)
    int drow[8];   // dst_node*32 + n  (add 16 for the second n-tile)
    if (full) {
      const int4* dp = (const int4*)(dst + base) + half * 2;
      int4 q0 = dp[0], q1 = dp[1];
      drow[0] = q0.x * 32 + n; drow[1] = q0.y * 32 + n;
      drow[2] = q0.z * 32 + n; drow[3] = q0.w * 32 + n;
      drow[4] = q1.x * 32 + n; drow[5] = q1.y * 32 + n;
      drow[6] = q1.z * 32 + n; drow[7] = q1.w * 32 + n;
    } else {
#pragma unroll
      for (int v = 0; v < 8; ++v) {
        int idx = base + half * 8 + v;
        drow[v] = ((idx < E) ? dst[idx] : 0) * 32 + n;
      }
    }

    // fp32-accurate product via hi/lo split: AhBh + AlBh + AhBl (per K-chunk)
    v8f c0 = {bias0, bias0, bias0, bias0, bias0, bias0, bias0, bias0};
    c0 = wmma_f16(a0h, Bhi[0][0], c0);
    c0 = wmma_f16(a0l, Bhi[0][0], c0);
    c0 = wmma_f16(a0h, Blo[0][0], c0);
    c0 = wmma_f16(a1h, Bhi[1][0], c0);
    c0 = wmma_f16(a1l, Bhi[1][0], c0);
    c0 = wmma_f16(a1h, Blo[1][0], c0);

    v8f c1 = {bias1, bias1, bias1, bias1, bias1, bias1, bias1, bias1};
    c1 = wmma_f16(a0h, Bhi[0][1], c1);
    c1 = wmma_f16(a0l, Bhi[0][1], c1);
    c1 = wmma_f16(a0h, Blo[0][1], c1);
    c1 = wmma_f16(a1h, Bhi[1][1], c1);
    c1 = wmma_f16(a1l, Bhi[1][1], c1);
    c1 = wmma_f16(a1h, Blo[1][1], c1);

    if (full) {
      // fast path: no per-row guards, no exec-mask churn
#pragma unroll
      for (int v = 0; v < 8; ++v) {
        float z0 = relu1(c0[v]);
        float z1 = relu1(c1[v]);
        atomicAdd(&acc[drow[v]], z0);
        atomicAdd(&acc[drow[v] + 16], z1);
        sumz0 += z0; ssq0 += z0 * z0;
        sumz1 += z1; ssq1 += z1 * z1;
      }
    } else {
#pragma unroll
      for (int v = 0; v < 8; ++v) {
        if (base + half * 8 + v < E) {
          float z0 = relu1(c0[v]);
          float z1 = relu1(c1[v]);
          atomicAdd(&acc[drow[v]], z0);
          atomicAdd(&acc[drow[v] + 16], z1);
          sumz0 += z0; ssq0 += z0 * z0;
          sumz1 += z1; ssq1 += z1 * z1;
        }
      }
    }
  }

  atomicAdd(&gsum[n],      sumz0);
  atomicAdd(&gssq[n],      ssq0);
  atomicAdd(&gsum[16 + n], sumz1);
  atomicAdd(&gssq[16 + n], ssq1);
}

// ---------------------------------------------------------------------------
// Per-node finalize: hidOut = cnt>0 ? (acc/cnt - mu)*g*rsqrt(var+eps)+beta : 0
// ---------------------------------------------------------------------------
__global__ void node_finalize_kernel(const float* __restrict__ acc,
                                     const float* __restrict__ cnt,
                                     const float* __restrict__ gsum,
                                     const float* __restrict__ gssq,
                                     const float* __restrict__ gamma,
                                     const float* __restrict__ beta,
                                     float* __restrict__ hidOut, int Nn,
                                     float invE) {
  int i = blockIdx.x * blockDim.x + threadIdx.x;
  if (i >= Nn * 32) return;
  int ch = i & 31;
  int node = i >> 5;
  float mu = gsum[ch] * invE;
  float var = gssq[ch] * invE - mu * mu;
  float scale = gamma[ch] * rsqrtf(var + EPS);
  float c = cnt[node];
  float out = 0.f;
  if (c > 0.f) out = (acc[i] / c - mu) * scale + beta[ch];
  hidOut[i] = out;
}

// ---------------------------------------------------------------------------
// Head part 1: t = relu(hid2 @ W3 + b3) [N,16]; accumulate batch stats
// ---------------------------------------------------------------------------
__global__ void head1_kernel(const float* __restrict__ hid2,
                             const float* __restrict__ W3,  // [32][16]
                             const float* __restrict__ b3,  // [16]
                             float* __restrict__ t3, float* __restrict__ sum3,
                             float* __restrict__ ssq3, int Nn) {
  int i = blockIdx.x * blockDim.x + threadIdx.x;
  if (i >= Nn * 16) return;
  int j = i & 15;
  int node = i >> 4;
  const float* h = hid2 + node * 32;
  float s = b3[j];
#pragma unroll
  for (int c = 0; c < 32; ++c) s += h[c] * W3[c * 16 + j];
  s = relu1(s);
  t3[i] = s;
  atomicAdd(&sum3[j], s);
  atomicAdd(&ssq3[j], s * s);
}

// ---------------------------------------------------------------------------
// Head part 2: BN over N, Lin(16,1), sigmoid
// ---------------------------------------------------------------------------
__global__ void head2_kernel(const float* __restrict__ t3,
                             const float* __restrict__ sum3,
                             const float* __restrict__ ssq3,
                             const float* __restrict__ g3,
                             const float* __restrict__ be3,
                             const float* __restrict__ W4,  // [16]
                             const float* __restrict__ b4,  // [1]
                             float* __restrict__ out, int Nn) {
  int node = blockIdx.x * blockDim.x + threadIdx.x;
  if (node >= Nn) return;
  float invN = 1.0f / (float)Nn;
  float y = b4[0];
#pragma unroll
  for (int j = 0; j < 16; ++j) {
    float mu = sum3[j] * invN;
    float var = ssq3[j] * invN - mu * mu;
    float x = (t3[node * 16 + j] - mu) * rsqrtf(var + EPS) * g3[j] + be3[j];
    y += x * W4[j];
  }
  out[node] = 1.0f / (1.0f + expf(-y));
}

// ---------------------------------------------------------------------------
extern "C" void kernel_launch(void* const* d_in, const int* in_sizes, int n_in,
                              void* d_out, int out_size, void* d_ws,
                              size_t ws_size, hipStream_t stream) {
  const float* feat = (const float*)d_in[0];
  const int*   src  = (const int*)d_in[1];
  const int*   dst  = (const int*)d_in[2];
  const float* W0 = (const float*)d_in[3];
  const float* b0 = (const float*)d_in[4];
  const float* W1 = (const float*)d_in[5];
  const float* b1 = (const float*)d_in[6];
  const float* g1 = (const float*)d_in[7];
  const float* be1 = (const float*)d_in[8];
  const float* W2 = (const float*)d_in[9];
  const float* b2 = (const float*)d_in[10];
  const float* g2 = (const float*)d_in[11];
  const float* be2 = (const float*)d_in[12];
  const float* W3 = (const float*)d_in[13];
  const float* b3 = (const float*)d_in[14];
  const float* g3 = (const float*)d_in[15];
  const float* be3 = (const float*)d_in[16];
  const float* W4 = (const float*)d_in[17];
  const float* b4 = (const float*)d_in[18];

  const int N = in_sizes[0] / 16;  // feat is [N, 8, 2]
  const int E = in_sizes[1];

  // ---- workspace layout (floats) ----
  float* ws = (float*)d_ws;
  size_t off = 0;
  float* cnt = ws + off;  off += (size_t)((N + 3) & ~3);
  float* accA = ws + off; off += 32ull * N;
  float* accB = ws + off; off += 32ull * N;
  float* stats = ws + off; off += 160;  // gsum1,gssq1,gsum2,gssq2,sum3,ssq3
  const size_t zeroFloats = off;
  float* hid0 = ws + off; off += 32ull * N;
  float* hid1 = ws + off; off += 32ull * N;
  float* hid2 = ws + off; off += 32ull * N;
  float* t3   = ws + off; off += 16ull * N;
  (void)ws_size;

  float* gsum1 = stats;       float* gssq1 = stats + 32;
  float* gsum2 = stats + 64;  float* gssq2 = stats + 96;
  float* sum3  = stats + 128; float* ssq3  = stats + 144;

  hipMemsetAsync(d_ws, 0, zeroFloats * sizeof(float), stream);

  const int TB = 256;
  embed_kernel<<<(N * 32 + TB - 1) / TB, TB, 0, stream>>>(feat, W0, b0, hid0, N);
  count_kernel<<<(E + TB - 1) / TB, TB, 0, stream>>>(dst, cnt, E);

  const float invE = 1.0f / (float)E;
  const int EDGE_BLOCKS = 1024;

  edgeconv_kernel<<<EDGE_BLOCKS, TB, 0, stream>>>(hid0, src, dst, W1, b1, accA,
                                                  gsum1, gssq1, E);
  node_finalize_kernel<<<(N * 32 + TB - 1) / TB, TB, 0, stream>>>(
      accA, cnt, gsum1, gssq1, g1, be1, hid1, N, invE);

  edgeconv_kernel<<<EDGE_BLOCKS, TB, 0, stream>>>(hid1, src, dst, W2, b2, accB,
                                                  gsum2, gssq2, E);
  node_finalize_kernel<<<(N * 32 + TB - 1) / TB, TB, 0, stream>>>(
      accB, cnt, gsum2, gssq2, g2, be2, hid2, N, invE);

  head1_kernel<<<(N * 16 + TB - 1) / TB, TB, 0, stream>>>(hid2, W3, b3, t3,
                                                          sum3, ssq3, N);
  head2_kernel<<<(N + TB - 1) / TB, TB, 0, stream>>>(t3, sum3, ssq3, g3, be3,
                                                     W4, b4, (float*)d_out, N);
}
